// Seq2Seq_82463372083841
// MI455X (gfx1250) — matrix operand order repaired and checked by reading, not compiled, not run on hardware
//
#include <hip/hip_runtime.h>

// ---------------------------------------------------------------------------
// Seq2Seq (enc LSTM 200 steps + dec LSTM 150 steps) on gfx1250.
// bf16 WMMA (16x16x32) + async global->LDS staging (ASYNCcnt) + double buffer.
// ---------------------------------------------------------------------------

typedef __attribute__((ext_vector_type(16))) __bf16 v16bf;
typedef __attribute__((ext_vector_type(8)))  __bf16 v8bf;
typedef __attribute__((ext_vector_type(8)))  float  v8f;

#define B_    256
#define EMB_  256
#define HID_  512
#define VOC_  231
#define VOCP_ 240   // accumulated (15 col tiles)
#define VOCS_ 256   // storage rows of padded W2 (even async staging)
#define SEQ_  200
#define T_    150
#define G4_   2048  // 4*HID

static __device__ __forceinline__ unsigned short f2bf(float f) {
  unsigned int u = __float_as_uint(f);
  unsigned int r = u + 0x7FFFu + ((u >> 16) & 1u);   // round-to-nearest-even
  return (unsigned short)(r >> 16);
}
static __device__ __forceinline__ float sigmoidf_(float x) {
  return 1.0f / (1.0f + __expf(-x));
}
static __device__ __forceinline__ float tanhf_(float x) {
  return 1.0f - 2.0f / (__expf(2.0f * x) + 1.0f);
}

// ---------------------------------------------------------------------------
// CDNA5 async global->LDS (per-lane 16B, tracked by ASYNCcnt)
// ---------------------------------------------------------------------------
static __device__ __forceinline__ void async_b128(const unsigned short* lds_dst,
                                                  const unsigned short* gsrc) {
  unsigned lds_off = (unsigned)(unsigned long long)lds_dst;   // low 32b = LDS addr
  asm volatile("global_load_async_to_lds_b128 %0, %1, off"
               :: "v"(lds_off), "v"(gsrc)
               : "memory");
}
#define ASYNC_WAIT(n) asm volatile("s_wait_asynccnt " #n ::: "memory")

// ---------------------------------------------------------------------------
// Prep kernels
// ---------------------------------------------------------------------------
__global__ void k_cvt_bf16(const float* __restrict__ s, unsigned short* __restrict__ d, int n) {
  int i = blockIdx.x * 256 + threadIdx.x;
  if (i < n) d[i] = f2bf(s[i]);
}
__global__ void k_bias_sum(const float* __restrict__ a, const float* __restrict__ b,
                           float* __restrict__ o, int n) {
  int i = blockIdx.x * 256 + threadIdx.x;
  if (i < n) o[i] = a[i] + b[i];
}
__global__ void k_w2pad(const float* __restrict__ W2, unsigned short* __restrict__ d) {
  int i = blockIdx.x * 256 + threadIdx.x;
  if (i >= VOCS_ * HID_) return;
  int j = i >> 9, k = i & 511;
  d[i] = (j < VOC_) ? f2bf(W2[j * HID_ + k]) : (unsigned short)0;
}
__global__ void k_b2pad(const float* __restrict__ b2, float* __restrict__ d) {
  int i = threadIdx.x;
  if (i < VOCP_) d[i] = (i < VOC_) ? b2[i] : -1e30f;   // padded cols lose argmax
}
// x[s][b][j] = train[b][s][0]*W[j][0] + train[b][s][1]*W[j][1] + bias[j]  (bf16)
__global__ void k_prep_x(const float* __restrict__ train, const float* __restrict__ W,
                         const float* __restrict__ bias, unsigned short* __restrict__ xs) {
  int i = blockIdx.x * 256 + threadIdx.x;
  if (i >= SEQ_ * B_ * EMB_) return;
  int j = i & 255, rem = i >> 8, b = rem & 255, s = rem >> 8;
  float v = train[(b * SEQ_ + s) * 2 + 0] * W[j * 2 + 0]
          + train[(b * SEQ_ + s) * 2 + 1] * W[j * 2 + 1] + bias[j];
  xs[i] = f2bf(v);
}
__global__ void k_zero_state(float* __restrict__ c, unsigned short* __restrict__ hbf) {
  int i = blockIdx.x * 256 + threadIdx.x;
  if (i < B_ * HID_) { c[i] = 0.0f; hbf[i] = 0; }
}
__global__ void k_zero_out(float* __restrict__ o) {
  int i = blockIdx.x * 256 + threadIdx.x;
  if (i < B_ * VOC_) o[i] = 0.0f;
}
__global__ void k_init_inp(const int* __restrict__ target, int* __restrict__ inp) {
  int b = threadIdx.x;
  if (b < B_) inp[b] = target[b * T_];
}

// ---------------------------------------------------------------------------
// Fragment helpers (wave32; contiguous k order used by this code for both A and B)
// A 16x32: lane m = lane&15; elems t   = K[t + 16h]
// B 32x16: lane n = lane&15; elems t   = K[t + 16h]
// C/D 16x16 f32: lane n = lane&15; reg r -> row r + 8h
// ---------------------------------------------------------------------------
static __device__ __forceinline__ v16bf load_a_frag(const unsigned short* p) {
  v8bf lo = *(const v8bf*)p;
  v8bf hi = *(const v8bf*)(p + 8);
  return __builtin_shufflevector(lo, hi, 0,1,2,3,4,5,6,7,8,9,10,11,12,13,14,15);
}
static __device__ __forceinline__ v16bf load_b_frag(const unsigned short* p) {
  v8bf lo = *(const v8bf*)p;
  v8bf hi = *(const v8bf*)(p + 8);
  return __builtin_shufflevector(lo, hi, 0,1,2,3,4,5,6,7,8,9,10,11,12,13,14,15);
}

// ---------------------------------------------------------------------------
// Fused LSTM step: gates = [x|e] @ Wih^T + h @ Whh^T + bsum ; cell update.
// grid = 8 blocks (64 hidden cols) x 512 threads (16 waves).
// Async double-buffered staging: 8 async b128 per wave per K-chunk.
// ---------------------------------------------------------------------------
template<bool GATHER>
__global__ __launch_bounds__(512)
void k_lstm_step(const unsigned short* __restrict__ xsrc,   // GATHER ? emb(231x256) : x_s(256x256)
                 const int* __restrict__ inp,               // GATHER only
                 const unsigned short* __restrict__ hprev,  // 256x512 bf16
                 const unsigned short* __restrict__ Wih,    // 2048x256 bf16
                 const unsigned short* __restrict__ Whh,    // 2048x512 bf16
                 const float* __restrict__ bsum,            // 2048 (b_ih + b_hh)
                 float* __restrict__ cst,                   // 256x512 f32 (in/out)
                 unsigned short* __restrict__ hout)         // 256x512 bf16 (new h)
{
  __shared__ alignas(16) unsigned short Alds[2][256 * 64];
  __shared__ alignas(16) unsigned short Blds[2][256 * 64];
  const int tid  = threadIdx.x;
  const int wave = tid >> 5;
  const int lane = tid & 31;
  const int hf   = lane >> 4;
  const int ln   = lane & 15;
  const int c0   = blockIdx.x * 64;   // hidden-unit column chunk

  auto stage = [&](int kc, int buf) {
    const int kbase = kc * 64;
    #pragma unroll
    for (int it = 0; it < 4; ++it) {              // A: 2048 x b128
      int idx = it * 512 + tid;
      int row = idx >> 3;
      int ko  = (idx & 7) << 3;
      const unsigned short* src;
      if (kbase < EMB_) {
        int ar = GATHER ? inp[row] : row;
        src = xsrc + ar * EMB_ + kbase + ko;
      } else {
        src = hprev + row * HID_ + (kbase - EMB_) + ko;
      }
      async_b128(&Alds[buf][idx << 3], src);
    }
    #pragma unroll
    for (int it = 0; it < 4; ++it) {              // B: 2048 x b128
      int idx = it * 512 + tid;
      int n   = idx >> 3;                         // gate-col 0..255
      int ko  = (idx & 7) << 3;
      int j   = (n >> 6) * HID_ + c0 + (n & 63);
      const unsigned short* src = (kbase < EMB_) ? (Wih + j * EMB_ + kbase + ko)
                                                 : (Whh + j * HID_ + (kbase - EMB_) + ko);
      async_b128(&Blds[buf][idx << 3], src);
    }
  };

  v8f acc[16];
  #pragma unroll
  for (int n = 0; n < 16; ++n) {
    float bv = bsum[(n >> 2) * HID_ + c0 + (n & 3) * 16 + ln];
    #pragma unroll
    for (int r = 0; r < 8; ++r) acc[n][r] = bv;
  }

  const int NKC = (EMB_ + HID_) / 64;   // 12
  stage(0, 0);
  for (int kc = 0; kc < NKC; ++kc) {
    const int buf = kc & 1;
    if (kc + 1 < NKC) {
      __syncthreads();                  // buf^1 free of readers (chunk kc-1 done)
      stage(kc + 1, buf ^ 1);
      ASYNC_WAIT(8);                    // our chunk kc (8 older async ops) landed
    } else {
      ASYNC_WAIT(0);
    }
    __syncthreads();                    // everyone's chunk kc visible
    #pragma unroll
    for (int kk = 0; kk < 64; kk += 32) {
      v16bf af = load_a_frag(&Alds[buf][(wave * 16 + ln) * 64 + kk + 16 * hf]);
      #pragma unroll
      for (int n = 0; n < 16; ++n) {
        v16bf bfr = load_b_frag(&Blds[buf][(n * 16 + ln) * 64 + kk + 16 * hf]);
        acc[n] = __builtin_amdgcn_wmma_f32_16x16x32_bf16(
            false, af, false, bfr, (short)0, acc[n], false, false);
      }
    }
  }

  // Elementwise LSTM cell fully in registers (tiles q,q+4,q+8,q+12 share cols)
  #pragma unroll
  for (int q = 0; q < 4; ++q) {
    int u = c0 + q * 16 + ln;
    #pragma unroll
    for (int r = 0; r < 8; ++r) {
      int b   = wave * 16 + hf * 8 + r;
      int off = b * HID_ + u;
      float iv = sigmoidf_(acc[q][r]);
      float fv = sigmoidf_(acc[4 + q][r]);
      float gv = tanhf_(acc[8 + q][r]);
      float ov = sigmoidf_(acc[12 + q][r]);
      float cn = fv * cst[off] + iv * gv;
      cst[off] = cn;
      hout[off] = f2bf(ov * tanhf_(cn));
    }
  }
}

// ---------------------------------------------------------------------------
// o1 = relu(h @ W1^T + b1), bf16 out. grid = 8 blocks x 512 threads.
// 5 async b128 per wave per K-chunk (A:4, B:1).
// ---------------------------------------------------------------------------
__global__ __launch_bounds__(512)
void k_fc_relu(const unsigned short* __restrict__ A,   // 256 x 512 bf16
               const unsigned short* __restrict__ W,   // 512 x 512 bf16
               const float* __restrict__ bias,
               unsigned short* __restrict__ outb)      // 256 x 512 bf16
{
  __shared__ alignas(16) unsigned short Alds[2][256 * 64];
  __shared__ alignas(16) unsigned short Blds[2][64 * 64];
  const int tid = threadIdx.x, wave = tid >> 5, lane = tid & 31;
  const int hf = lane >> 4, ln = lane & 15;
  const int c0 = blockIdx.x * 64;

  auto stage = [&](int kc, int buf) {
    const int kbase = kc * 64;
    #pragma unroll
    for (int it = 0; it < 4; ++it) {
      int idx = it * 512 + tid;
      int row = idx >> 3, ko = (idx & 7) << 3;
      async_b128(&Alds[buf][idx << 3], A + row * HID_ + kbase + ko);
    }
    {
      int idx = tid;                       // 512 x b128 covers 64x64
      int n = idx >> 3, ko = (idx & 7) << 3;
      async_b128(&Blds[buf][idx << 3], W + (c0 + n) * HID_ + kbase + ko);
    }
  };

  v8f acc[4];
  #pragma unroll
  for (int n = 0; n < 4; ++n) {
    float bv = bias[c0 + n * 16 + ln];
    #pragma unroll
    for (int r = 0; r < 8; ++r) acc[n][r] = bv;
  }
  const int NKC = HID_ / 64;   // 8
  stage(0, 0);
  for (int kc = 0; kc < NKC; ++kc) {
    const int buf = kc & 1;
    if (kc + 1 < NKC) {
      __syncthreads();
      stage(kc + 1, buf ^ 1);
      ASYNC_WAIT(5);
    } else {
      ASYNC_WAIT(0);
    }
    __syncthreads();
    #pragma unroll
    for (int kk = 0; kk < 64; kk += 32) {
      v16bf af = load_a_frag(&Alds[buf][(wave * 16 + ln) * 64 + kk + 16 * hf]);
      #pragma unroll
      for (int n = 0; n < 4; ++n) {
        v16bf bfr = load_b_frag(&Blds[buf][(n * 16 + ln) * 64 + kk + 16 * hf]);
        acc[n] = __builtin_amdgcn_wmma_f32_16x16x32_bf16(
            false, af, false, bfr, (short)0, acc[n], false, false);
      }
    }
  }
  #pragma unroll
  for (int n = 0; n < 4; ++n) {
    int col = c0 + n * 16 + ln;
    #pragma unroll
    for (int r = 0; r < 8; ++r) {
      int b = wave * 16 + hf * 8 + r;
      float v = acc[n][r];
      outb[b * HID_ + col] = f2bf(v > 0.0f ? v : 0.0f);
    }
  }
}

// ---------------------------------------------------------------------------
// logits = relu(o1 @ W2p^T + b2p) (240 cols acc, W2 stored padded to 256 rows);
// write 231 cols; per-row argmax -> inp. grid = 1 block x 512 threads.
// 8 async b128 per wave per K-chunk.
// ---------------------------------------------------------------------------
__global__ __launch_bounds__(512)
void k_out2_argmax(const unsigned short* __restrict__ A,    // 256 x 512 bf16
                   const unsigned short* __restrict__ W2p,  // 256 x 512 bf16 (rows>=231 zero)
                   const float* __restrict__ b2p,           // 240 (pads = -1e30)
                   float* __restrict__ out,                 // 256 x 231 (this step)
                   int* __restrict__ inp)                   // 256 next tokens
{
  __shared__ alignas(16) unsigned short Alds[2][256 * 64];
  __shared__ alignas(16) unsigned short Blds[2][256 * 64];
  const int tid = threadIdx.x, wave = tid >> 5, lane = tid & 31;
  const int hf = lane >> 4, ln = lane & 15;

  auto stage = [&](int kc, int buf) {
    const int kbase = kc * 64;
    #pragma unroll
    for (int it = 0; it < 4; ++it) {
      int idx = it * 512 + tid;
      int row = idx >> 3, ko = (idx & 7) << 3;
      async_b128(&Alds[buf][idx << 3], A + row * HID_ + kbase + ko);
    }
    #pragma unroll
    for (int it = 0; it < 4; ++it) {
      int idx = it * 512 + tid;
      int n = idx >> 3, ko = (idx & 7) << 3;
      async_b128(&Blds[buf][idx << 3], W2p + n * HID_ + kbase + ko);
    }
  };

  v8f acc[15];
  #pragma unroll
  for (int n = 0; n < 15; ++n) {
    float bv = b2p[n * 16 + ln];
    #pragma unroll
    for (int r = 0; r < 8; ++r) acc[n][r] = bv;
  }
  const int NKC = HID_ / 64;   // 8
  stage(0, 0);
  for (int kc = 0; kc < NKC; ++kc) {
    const int buf = kc & 1;
    if (kc + 1 < NKC) {
      __syncthreads();
      stage(kc + 1, buf ^ 1);
      ASYNC_WAIT(8);
    } else {
      ASYNC_WAIT(0);
    }
    __syncthreads();
    #pragma unroll
    for (int kk = 0; kk < 64; kk += 32) {
      v16bf af = load_a_frag(&Alds[buf][(wave * 16 + ln) * 64 + kk + 16 * hf]);
      #pragma unroll
      for (int n = 0; n < 15; ++n) {
        v16bf bfr = load_b_frag(&Blds[buf][(n * 16 + ln) * 64 + kk + 16 * hf]);
        acc[n] = __builtin_amdgcn_wmma_f32_16x16x32_bf16(
            false, af, false, bfr, (short)0, acc[n], false, false);
      }
    }
  }
  // ReLU + store + argmax (tie-break: lowest index, matching jnp.argmax)
  #pragma unroll
  for (int r = 0; r < 8; ++r) {
    int b = wave * 16 + hf * 8 + r;
    float bv = -1.0f; int bi = 0;
    #pragma unroll
    for (int n = 0; n < 15; ++n) {
      int col = n * 16 + ln;
      float v = acc[n][r];
      v = v > 0.0f ? v : 0.0f;
      if (col < VOC_) out[b * VOC_ + col] = v;
      if (v > bv) { bv = v; bi = col; }
    }
    #pragma unroll
    for (int m = 1; m < 16; m <<= 1) {
      float ov = __shfl_xor(bv, m, 16);
      int   oi = __shfl_xor(bi, m, 16);
      if (ov > bv || (ov == bv && oi < bi)) { bv = ov; bi = oi; }
    }
    if (ln == 0) inp[b] = bi;
  }
}

// ---------------------------------------------------------------------------
extern "C" void kernel_launch(void* const* d_in, const int* in_sizes, int n_in,
                              void* d_out, int out_size, void* d_ws, size_t ws_size,
                              hipStream_t stream) {
  (void)in_sizes; (void)n_in; (void)out_size; (void)ws_size;
  const float* train  = (const float*)d_in[0];
  const int*   target = (const int*)  d_in[1];
  const float* efcW   = (const float*)d_in[2];
  const float* efcb   = (const float*)d_in[3];
  const float* eWih   = (const float*)d_in[4];
  const float* eWhh   = (const float*)d_in[5];
  const float* ebih   = (const float*)d_in[6];
  const float* ebhh   = (const float*)d_in[7];
  const float* embf   = (const float*)d_in[8];
  const float* dWih   = (const float*)d_in[9];
  const float* dWhh   = (const float*)d_in[10];
  const float* dbih   = (const float*)d_in[11];
  const float* dbhh   = (const float*)d_in[12];
  const float* W1     = (const float*)d_in[13];
  const float* b1     = (const float*)d_in[14];
  const float* W2     = (const float*)d_in[15];
  const float* b2     = (const float*)d_in[16];
  float* out = (float*)d_out;

  char* ws = (char*)d_ws;
  size_t cur = 0;
  auto alloc = [&](size_t bytes) -> char* {
    char* p = ws + cur;
    cur = (cur + bytes + 255) & ~(size_t)255;
    return p;
  };
  unsigned short* xs    = (unsigned short*)alloc((size_t)SEQ_ * B_ * EMB_ * 2);
  unsigned short* Wih_e = (unsigned short*)alloc((size_t)G4_ * EMB_ * 2);
  unsigned short* Whh_e = (unsigned short*)alloc((size_t)G4_ * HID_ * 2);
  unsigned short* Wih_d = (unsigned short*)alloc((size_t)G4_ * EMB_ * 2);
  unsigned short* Whh_d = (unsigned short*)alloc((size_t)G4_ * HID_ * 2);
  unsigned short* W1b   = (unsigned short*)alloc((size_t)HID_ * HID_ * 2);
  unsigned short* W2p   = (unsigned short*)alloc((size_t)VOCS_ * HID_ * 2);
  unsigned short* embb  = (unsigned short*)alloc((size_t)VOC_ * EMB_ * 2);
  float*          bsumE = (float*)alloc(G4_ * 4);
  float*          bsumD = (float*)alloc(G4_ * 4);
  float*          b2p   = (float*)alloc(VOCP_ * 4);
  float*          cbuf  = (float*)alloc((size_t)B_ * HID_ * 4);
  unsigned short* hbf0  = (unsigned short*)alloc((size_t)B_ * HID_ * 2);
  unsigned short* hbf1  = (unsigned short*)alloc((size_t)B_ * HID_ * 2);
  unsigned short* o1b   = (unsigned short*)alloc((size_t)B_ * HID_ * 2);
  int*            inp   = (int*)alloc(B_ * 4);

  auto cvt = [&](const float* s, unsigned short* d, int n) {
    k_cvt_bf16<<<(n + 255) / 256, 256, 0, stream>>>(s, d, n);
  };
  cvt(eWih, Wih_e, G4_ * EMB_);
  cvt(eWhh, Whh_e, G4_ * HID_);
  cvt(dWih, Wih_d, G4_ * EMB_);
  cvt(dWhh, Whh_d, G4_ * HID_);
  cvt(W1,   W1b,   HID_ * HID_);
  cvt(embf, embb,  VOC_ * EMB_);
  k_w2pad<<<(VOCS_ * HID_ + 255) / 256, 256, 0, stream>>>(W2, W2p);
  k_bias_sum<<<G4_ / 256, 256, 0, stream>>>(ebih, ebhh, bsumE, G4_);
  k_bias_sum<<<G4_ / 256, 256, 0, stream>>>(dbih, dbhh, bsumD, G4_);
  k_b2pad<<<1, 256, 0, stream>>>(b2, b2p);
  k_prep_x<<<(SEQ_ * B_ * EMB_ + 255) / 256, 256, 0, stream>>>(train, efcW, efcb, xs);
  k_zero_state<<<(B_ * HID_ + 255) / 256, 256, 0, stream>>>(cbuf, hbf0);
  k_zero_out<<<(B_ * VOC_ + 255) / 256, 256, 0, stream>>>(out);
  k_init_inp<<<1, 256, 0, stream>>>(target, inp);

  unsigned short* hA = hbf0;
  unsigned short* hB = hbf1;
  // Encoder: 200 sequential fused steps (h ping-pong across launches)
  for (int s = 0; s < SEQ_; ++s) {
    k_lstm_step<false><<<8, 512, 0, stream>>>(xs + (size_t)s * B_ * EMB_, nullptr,
                                              hA, Wih_e, Whh_e, bsumE, cbuf, hB);
    unsigned short* t = hA; hA = hB; hB = t;
  }
  // Decoder: 149 sequential steps: gates+cell -> out1 -> out2+argmax
  for (int t = 1; t < T_; ++t) {
    k_lstm_step<true><<<8, 512, 0, stream>>>(embb, inp, hA, Wih_d, Whh_d, bsumD, cbuf, hB);
    k_fc_relu<<<8, 512, 0, stream>>>(hB, W1b, b1, o1b);
    k_out2_argmax<<<1, 512, 0, stream>>>(o1b, W2p, b2p, out + (size_t)t * B_ * VOC_, inp);
    unsigned short* tmp = hA; hA = hB; hB = tmp;
  }
}
